// Seq2seqLSTMWithAttention_55551107006808
// MI455X (gfx1250) — compile-verified, matrix operand order checked by reading, not run on hardware
//
#include <hip/hip_runtime.h>
#include <hip/hip_bf16.h>

// ---------------------------------------------------------------------------
// Seq2seq LSTM + attention for MI455X (gfx1250, wave32, WMMA).
// B=64, S=512, F=256, H=512, HA=64, T=96.
// All GEMMs use v_wmma_f32_16x16x32_f16 (f16 in, f32 accum).
// Fragment loads are two 16-byte chunks at immediate offsets; GEMM inner
// loops are double-buffered AND pinned with sched_group_barrier so the
// scheduler keeps [VMEM x10][WMMA x4] groups interleaved (partial loadcnt
// waits instead of full drains; RA cannot coalesce the two buffer sets).
// ---------------------------------------------------------------------------

typedef _Float16 v16h  __attribute__((ext_vector_type(16)));
typedef float    v8f   __attribute__((ext_vector_type(8)));
typedef unsigned uint4v __attribute__((ext_vector_type(4)));

union Frag { v16h h; uint4v q[2]; };

__device__ __forceinline__ void ldfrag(Frag& f, const _Float16* p) {
  f.q[0] = *(const uint4v*)(p);        // halfs [0..7]   -> VGPR pairs r0..r3
  f.q[1] = *(const uint4v*)(p + 16);   // halfs [16..23] -> VGPR pairs r4..r7
}

__device__ __forceinline__ v8f wmma16(const Frag& a, const Frag& b, v8f c) {
  return __builtin_amdgcn_wmma_f32_16x16x32_f16(false, a.h, false, b.h,
                                                (short)0, c, false, false);
}

// scheduling-group masks (LLVM AMDGPU): 0x008 = MFMA/WMMA, 0x020 = VMEM read
#define SG_VMEM_RD(n) __builtin_amdgcn_sched_group_barrier(0x020, (n), 0)
#define SG_WMMA(n)    __builtin_amdgcn_sched_group_barrier(0x008, (n), 0)

// ------------------------- one-time conversion ------------------------------

__global__ void cvt_f32_f16(const float* __restrict__ s, _Float16* __restrict__ d, long n) {
  long i = (long)blockIdx.x * blockDim.x + threadIdx.x;
  long st = (long)gridDim.x * blockDim.x;
  for (; i < n; i += st) d[i] = (_Float16)s[i];
}

__global__ void init_state(const float* __restrict__ h0, const float* __restrict__ c0,
                           const float* __restrict__ iseq, const float* __restrict__ Wa,
                           float* __restrict__ hf, float* __restrict__ cb,
                           _Float16* __restrict__ h16, _Float16* __restrict__ xdec,
                           _Float16* __restrict__ wae) {
  int i = blockIdx.x * 256 + threadIdx.x;            // grid covers 32768
  if (i < 64 * 512) {
    float h = h0[i];
    hf[i] = h; h16[i] = (_Float16)h; cb[i] = c0[i];
  }
  if (i < 64 * 256) {                                // x_dec0 = input_seq[:, S-1, :]
    int b = i >> 8, k = i & 255;
    xdec[i] = (_Float16)iseq[(size_t)b * 512 * 256 + 511 * 256 + k];
  }
  if (i < 64 * 64) {                                 // Wa_e = Wa[:, HA:]
    int n = i >> 6, k = i & 63;
    wae[i] = (_Float16)Wa[n * 128 + 64 + k];
  }
}

// ------------------------- fused LSTM step (enc & dec) ----------------------
// grid = 16 WGs (j-slice of 32), block = 256 (8 waves).
// gates(64 x 2048) = [x|h](64 x 768) @ W^T, W = [Wih|Whh] rows = gate index.
// Wave w: M-tile mt = w&3, gate-half nh = w>>2 (gates {i,f} or {g,o}).

__global__ __launch_bounds__(256)
void lstm_step_k(const _Float16* __restrict__ x16, int xstride,
                 const _Float16* __restrict__ h16in,
                 const _Float16* __restrict__ Wih, const _Float16* __restrict__ Whh,
                 const float* __restrict__ bias,
                 float* __restrict__ cbuf, float* __restrict__ hf32,
                 _Float16* __restrict__ h16out, _Float16* __restrict__ encst) {
  __shared__ float gl[64 * 128];
  const int tid = threadIdx.x, lane = tid & 31, w = tid >> 5;
  const int mt = w & 3, nh = w >> 2, mb = mt * 16;
  const int J = blockIdx.x * 32;
  const int l16 = lane & 15, g = lane >> 4;
  const int arow = mb + l16;
  const int koff = 8 * g;

  const _Float16* aX = x16 + (size_t)arow * xstride + koff;
  const _Float16* aH = h16in + arow * 512 + koff;
  const _Float16* bX[4];
  const _Float16* bH[4];
#pragma unroll
  for (int i = 0; i < 4; ++i) {
    int nt = nh * 4 + i;                             // gate q = nt>>1
    int brow = (nt >> 1) * 512 + J + (nt & 1) * 16 + l16;
    bX[i] = Wih + brow * 256 + koff;
    bH[i] = Whh + brow * 512 + koff;
  }
  __builtin_prefetch(bH[0], 0, 1);                   // pull h-phase weights toward L0
  __builtin_prefetch(bH[2], 0, 1);

  // unified 24-tile K pipeline: kt<8 -> x/Wih region, kt>=8 -> h/Whh region.
  // Region select folds at compile time under full unroll.
  auto ap = [&](int kt) -> const _Float16* {
    return (kt < 8) ? (aX + kt * 32) : (aH + (kt - 8) * 32);
  };
  auto bp = [&](int i, int kt) -> const _Float16* {
    return (kt < 8) ? (bX[i] + kt * 32) : (bH[i] + (kt - 8) * 32);
  };

  v8f acc[4] = {};
  Frag aE, aO, bE[4], bO[4];
  ldfrag(aE, ap(0));
#pragma unroll
  for (int i = 0; i < 4; ++i) ldfrag(bE[i], bp(i, 0));
  SG_VMEM_RD(10);                                    // prologue loads first
#pragma unroll
  for (int kt = 0; kt < 24; kt += 2) {
    ldfrag(aO, ap(kt + 1));                          // load odd set ahead
#pragma unroll
    for (int i = 0; i < 4; ++i) ldfrag(bO[i], bp(i, kt + 1));
#pragma unroll
    for (int i = 0; i < 4; ++i) acc[i] = wmma16(aE, bE[i], acc[i]);
    if (kt + 2 < 24) {
      ldfrag(aE, ap(kt + 2));                        // reload even set ahead
#pragma unroll
      for (int i = 0; i < 4; ++i) ldfrag(bE[i], bp(i, kt + 2));
    }
#pragma unroll
    for (int i = 0; i < 4; ++i) acc[i] = wmma16(aO, bO[i], acc[i]);
    // enforce static order: [VMEM 10][WMMA 4][VMEM 10][WMMA 4] per K-pair
    SG_VMEM_RD(10);
    SG_WMMA(4);
    if (kt + 2 < 24) SG_VMEM_RD(10);
    SG_WMMA(4);
  }

#pragma unroll
  for (int i = 0; i < 4; ++i) {
    int col = (nh * 4 + i) * 16 + l16;               // == q*32 + half*16 + l16
#pragma unroll
    for (int r = 0; r < 8; ++r) {
      int m = mb + r + 8 * g;
      gl[m * 128 + col] = acc[i][r];
    }
  }
  __syncthreads();
  // pointwise LSTM update: 64 batches x 32 j's = 2048 elems, 8 per thread
#pragma unroll
  for (int it = 0; it < 8; ++it) {
    int e = tid + it * 256;
    int m = e >> 5, jl = e & 31, j = J + jl;
    float iv = gl[m * 128 + jl]        + bias[j];
    float fv = gl[m * 128 + 32 + jl]   + bias[512 + j];
    float gv = gl[m * 128 + 64 + jl]   + bias[1024 + j];
    float ov = gl[m * 128 + 96 + jl]   + bias[1536 + j];
    iv = 1.f / (1.f + __expf(-iv));
    fv = 1.f / (1.f + __expf(-fv));
    ov = 1.f / (1.f + __expf(-ov));
    float cc = fv * cbuf[m * 512 + j] + iv * tanhf(gv);
    float hh = ov * tanhf(cc);
    cbuf[m * 512 + j] = cc;
    hf32[m * 512 + j] = hh;
    _Float16 h16v = (_Float16)hh;
    h16out[m * 512 + j] = h16v;
    if (encst) encst[(size_t)m * (512 * 512) + j] = h16v;  // enc_out[b][t][j]
  }
}

// ---------------- enc_proj + enc_attn (fused two-stage WMMA GEMM) -----------
// M = B*S = 32768 rows. grid = 256 WGs x 8 waves; wave owns one 16-row M-tile.

__global__ __launch_bounds__(256)
void proj_attn_k(const _Float16* __restrict__ enc16, const _Float16* __restrict__ Wep,
                 const float* __restrict__ bep, const _Float16* __restrict__ Wae,
                 float* __restrict__ encP, float* __restrict__ encA) {
  __shared__ __align__(16) _Float16 ptile[8 * 16 * 64];
  const int tid = threadIdx.x, lane = tid & 31, w = tid >> 5;
  const int mtile = blockIdx.x * 8 + w, mb = mtile * 16;
  const int l16 = lane & 15, g = lane >> 4;
  const int koff = 8 * g;

  const _Float16* aB = enc16 + (size_t)(mb + l16) * 512 + koff;
  const _Float16* bB[4];
#pragma unroll
  for (int i = 0; i < 4; ++i) bB[i] = Wep + (i * 16 + l16) * 512 + koff;

  v8f acc[4] = {};
  Frag aE, aO, bE[4], bO[4];
  ldfrag(aE, aB);
#pragma unroll
  for (int i = 0; i < 4; ++i) ldfrag(bE[i], bB[i]);
  SG_VMEM_RD(10);
#pragma unroll
  for (int kt = 0; kt < 16; kt += 2) {
    ldfrag(aO, aB + (kt + 1) * 32);
#pragma unroll
    for (int i = 0; i < 4; ++i) ldfrag(bO[i], bB[i] + (kt + 1) * 32);
#pragma unroll
    for (int i = 0; i < 4; ++i) acc[i] = wmma16(aE, bE[i], acc[i]);
    if (kt + 2 < 16) {
      ldfrag(aE, aB + (kt + 2) * 32);
#pragma unroll
      for (int i = 0; i < 4; ++i) ldfrag(bE[i], bB[i] + (kt + 2) * 32);
    }
#pragma unroll
    for (int i = 0; i < 4; ++i) acc[i] = wmma16(aO, bO[i], acc[i]);
    SG_VMEM_RD(10);
    SG_WMMA(4);
    if (kt + 2 < 16) SG_VMEM_RD(10);
    SG_WMMA(4);
  }

  _Float16* pt = ptile + w * (16 * 64);
#pragma unroll
  for (int i = 0; i < 4; ++i) {
    int n = i * 16 + l16;
    float bb = bep[n];
#pragma unroll
    for (int r = 0; r < 8; ++r) {
      int ml = r + 8 * g;
      float vv = acc[i][r] + bb;
      encP[(size_t)(mb + ml) * 64 + n] = vv;
      pt[ml * 64 + n] = (_Float16)vv;                // lane-transpose via LDS
    }
  }
  __syncthreads();
  const _Float16* aL = pt + l16 * 64 + koff;         // LDS A fragments
  const _Float16* bA[4];
#pragma unroll
  for (int i = 0; i < 4; ++i) bA[i] = Wae + (i * 16 + l16) * 64 + koff;
  v8f acc2[4] = {};
#pragma unroll
  for (int kt = 0; kt < 2; ++kt) {
    Frag a; ldfrag(a, aL + kt * 32);
    Frag b0, b1, b2, b3;
    ldfrag(b0, bA[0] + kt * 32);
    ldfrag(b1, bA[1] + kt * 32);
    ldfrag(b2, bA[2] + kt * 32);
    ldfrag(b3, bA[3] + kt * 32);
    acc2[0] = wmma16(a, b0, acc2[0]);
    acc2[1] = wmma16(a, b1, acc2[1]);
    acc2[2] = wmma16(a, b2, acc2[2]);
    acc2[3] = wmma16(a, b3, acc2[3]);
  }
#pragma unroll
  for (int i = 0; i < 4; ++i)
#pragma unroll
    for (int r = 0; r < 8; ++r)
      encA[(size_t)(mb + r + 8 * g) * 64 + i * 16 + l16] = acc2[i][r];
}

// ---------------------- decoder attention (VALU, 1 WG / batch) --------------

__global__ __launch_bounds__(256)
void attn_step_k(const float* __restrict__ hf, const float* __restrict__ Whp,
                 const float* __restrict__ bhp, const float* __restrict__ Wa,
                 const float* __restrict__ ba, const float* __restrict__ vv,
                 const float* __restrict__ encP, const float* __restrict__ encA,
                 float* __restrict__ ctx32, _Float16* __restrict__ ctx16) {
  __shared__ float hq[64], e2[64], sc[512], red[256], part[256];
  int b = blockIdx.x, tid = threadIdx.x;
  if (tid < 64) {
    float s = bhp[tid];
    const float* hb = hf + b * 512;
    const float* wr = Whp + tid * 512;
    for (int j = 0; j < 512; ++j) s += hb[j] * wr[j];
    hq[tid] = s;
  }
  __syncthreads();
  if (tid < 64) {
    float s = ba[tid];
    const float* wr = Wa + tid * 128;                // Wa_h = Wa[:, :HA]
    for (int k = 0; k < 64; ++k) s += hq[k] * wr[k];
    e2[tid] = s;
  }
  __syncthreads();
  for (int s0 = tid; s0 < 512; s0 += 256) {
    const float* ea = encA + ((size_t)b * 512 + s0) * 64;
    float sco = 0.f;
    for (int h = 0; h < 64; ++h) sco += vv[h] * tanhf(e2[h] + ea[h]);
    sc[s0] = sco;
  }
  __syncthreads();
  float mx = fmaxf(sc[tid], sc[tid + 256]);
  red[tid] = mx; __syncthreads();
  for (int st = 128; st > 0; st >>= 1) {
    if (tid < st) red[tid] = fmaxf(red[tid], red[tid + st]);
    __syncthreads();
  }
  mx = red[0]; __syncthreads();
  float p0 = __expf(sc[tid] - mx), p1 = __expf(sc[tid + 256] - mx);
  sc[tid] = p0; sc[tid + 256] = p1;
  red[tid] = p0 + p1; __syncthreads();
  for (int st = 128; st > 0; st >>= 1) {
    if (tid < st) red[tid] += red[tid + st];
    __syncthreads();
  }
  float Z = red[0];
  int ha = tid & 63, ch = tid >> 6;
  float a = 0.f;
  const float* pb = encP + ((size_t)b * 512) * 64;
  for (int s0 = ch * 128; s0 < ch * 128 + 128; ++s0) a += sc[s0] * pb[s0 * 64 + ha];
  part[tid] = a; __syncthreads();
  if (tid < 64) {
    float c = (part[tid] + part[tid + 64] + part[tid + 128] + part[tid + 192]) / Z;
    ctx32[b * 64 + tid] = c;
    ctx16[b * 64 + tid] = (_Float16)c;
  }
}

// ---------------------- output projection (WMMA 64x256x576) -----------------
// grid = 8 WGs x 8 waves; wave owns one 16x16 tile of out(64 x 256).

__global__ __launch_bounds__(256)
void out_proj_k(const _Float16* __restrict__ h16, const _Float16* __restrict__ ctx16,
                const _Float16* __restrict__ Wo16, const float* __restrict__ bo,
                _Float16* __restrict__ xdec, float* __restrict__ dout, int t) {
  const int tid = threadIdx.x, lane = tid & 31, w = tid >> 5;
  const int idx = blockIdx.x * 8 + w;                // 0..63
  const int mt = idx >> 4, ntile = idx & 15;
  const int mb = mt * 16, l16 = lane & 15, g = lane >> 4;
  const int n = ntile * 16 + l16;
  const int koff = 8 * g;

  const _Float16* aH = h16 + (mb + l16) * 512 + koff;
  const _Float16* aC = ctx16 + (mb + l16) * 64 + koff;
  const _Float16* bB = Wo16 + n * 576 + koff;

  auto ap = [&](int kt) -> const _Float16* {
    return (kt < 16) ? (aH + kt * 32) : (aC + (kt - 16) * 32);
  };

  v8f acc = {};
  Frag aE, aO, bE, bO;
  ldfrag(aE, ap(0));
  ldfrag(bE, bB);
  SG_VMEM_RD(4);
#pragma unroll
  for (int kt = 0; kt < 18; kt += 2) {               // K = 576 = 18 tiles
    ldfrag(aO, ap(kt + 1));
    ldfrag(bO, bB + (kt + 1) * 32);
    acc = wmma16(aE, bE, acc);
    if (kt + 2 < 18) {
      ldfrag(aE, ap(kt + 2));
      ldfrag(bE, bB + (kt + 2) * 32);
    }
    acc = wmma16(aO, bO, acc);
    SG_VMEM_RD(4);
    SG_WMMA(1);
    if (kt + 2 < 18) SG_VMEM_RD(4);
    SG_WMMA(1);
  }
  float bb = bo[n];
#pragma unroll
  for (int r = 0; r < 8; ++r) {
    int m = mb + r + 8 * g;
    float vv = acc[r] + bb;
    xdec[m * 256 + n] = (_Float16)vv;                // feedback input for next step
    if (n == 255) dout[m * 96 + t] = vv;             // preds[:, :, -1].T
  }
}

// --------------------------------- host -------------------------------------

static inline size_t alignup(size_t x) { return (x + 255) & ~(size_t)255; }

extern "C" void kernel_launch(void* const* d_in, const int* in_sizes, int n_in,
                              void* d_out, int out_size, void* d_ws, size_t ws_size,
                              hipStream_t stream) {
  (void)in_sizes; (void)n_in; (void)out_size;
  const float* in_seq = (const float*)d_in[0];
  const float* h0     = (const float*)d_in[1];
  const float* c0     = (const float*)d_in[2];
  const float* Wih_e  = (const float*)d_in[3];
  const float* Whh_e  = (const float*)d_in[4];
  const float* b_e    = (const float*)d_in[5];
  const float* Wih_d  = (const float*)d_in[6];
  const float* Whh_d  = (const float*)d_in[7];
  const float* b_d    = (const float*)d_in[8];
  const float* We_p   = (const float*)d_in[9];
  const float* be_p   = (const float*)d_in[10];
  const float* Wh_p   = (const float*)d_in[11];
  const float* bh_p   = (const float*)d_in[12];
  const float* Wa     = (const float*)d_in[13];
  const float* ba     = (const float*)d_in[14];
  const float* v      = (const float*)d_in[15];
  const float* Wo     = (const float*)d_in[16];
  const float* bo     = (const float*)d_in[17];
  float* dout = (float*)d_out;

  char* ws = (char*)d_ws;
  size_t o = 0;
  auto take = [&](size_t bytes) { size_t r = o; o = alignup(o + bytes); return r; };
  _Float16* WIHE = (_Float16*)(ws + take((size_t)2048 * 256 * 2));
  _Float16* WHHE = (_Float16*)(ws + take((size_t)2048 * 512 * 2));
  _Float16* WIHD = (_Float16*)(ws + take((size_t)2048 * 256 * 2));
  _Float16* WHHD = (_Float16*)(ws + take((size_t)2048 * 512 * 2));
  _Float16* WEP  = (_Float16*)(ws + take((size_t)64 * 512 * 2));
  _Float16* WAE  = (_Float16*)(ws + take((size_t)64 * 64 * 2));
  _Float16* WO16 = (_Float16*)(ws + take((size_t)256 * 576 * 2));
  _Float16* IN16 = (_Float16*)(ws + take((size_t)64 * 512 * 256 * 2));
  _Float16* ENC16= (_Float16*)(ws + take((size_t)64 * 512 * 512 * 2));
  float*    ENCP = (float*)   (ws + take((size_t)64 * 512 * 64 * 4));
  float*    ENCA = (float*)   (ws + take((size_t)64 * 512 * 64 * 4));
  _Float16* H16A = (_Float16*)(ws + take((size_t)64 * 512 * 2));
  _Float16* H16B = (_Float16*)(ws + take((size_t)64 * 512 * 2));
  float*    HF32 = (float*)   (ws + take((size_t)64 * 512 * 4));
  float*    CF32 = (float*)   (ws + take((size_t)64 * 512 * 4));
  float*    CTX32= (float*)   (ws + take((size_t)64 * 64 * 4));
  _Float16* CTX16= (_Float16*)(ws + take((size_t)64 * 64 * 2));
  _Float16* XDEC = (_Float16*)(ws + take((size_t)64 * 256 * 2));
  if (o > ws_size) return;  // workspace too small; bail deterministically

  auto cvt = [&](const float* s, _Float16* d, long n) {
    int blocks = (int)((n + 255) / 256);
    if (blocks > 8192) blocks = 8192;
    cvt_f32_f16<<<blocks, 256, 0, stream>>>(s, d, n);
  };
  cvt(Wih_e, WIHE, (long)2048 * 256);
  cvt(Whh_e, WHHE, (long)2048 * 512);
  cvt(Wih_d, WIHD, (long)2048 * 256);
  cvt(Whh_d, WHHD, (long)2048 * 512);
  cvt(We_p,  WEP,  (long)64 * 512);
  cvt(Wo,    WO16, (long)256 * 576);
  cvt(in_seq, IN16, (long)64 * 512 * 256);
  init_state<<<128, 256, 0, stream>>>(h0, c0, in_seq, Wa, HF32, CF32, H16A, XDEC, WAE);

  // ---------------- encoder: 512 recurrent steps -----------------
  for (int t = 0; t < 512; ++t) {
    const _Float16* hin = (t & 1) ? H16B : H16A;
    _Float16* hout      = (t & 1) ? H16A : H16B;
    lstm_step_k<<<16, 256, 0, stream>>>(IN16 + (size_t)t * 256, 512 * 256, hin,
                                        WIHE, WHHE, b_e, CF32, HF32, hout,
                                        ENC16 + (size_t)t * 512);
  }

  // ---------------- enc_proj + enc_attn (one big WMMA GEMM) ------
  proj_attn_k<<<256, 256, 0, stream>>>(ENC16, WEP, be_p, WAE, ENCP, ENCA);

  // ---------------- decoder: 96 recurrent steps ------------------
  for (int d = 0; d < 96; ++d) {
    int step = 512 + d;
    const _Float16* hin = (step & 1) ? H16B : H16A;
    _Float16* hout      = (step & 1) ? H16A : H16B;
    attn_step_k<<<64, 256, 0, stream>>>(HF32, Wh_p, bh_p, Wa, ba, v,
                                        ENCP, ENCA, CTX32, CTX16);
    lstm_step_k<<<16, 256, 0, stream>>>(XDEC, 256, hin, WIHD, WHHD, b_d,
                                        CF32, HF32, hout, (_Float16*)nullptr);
    out_proj_k<<<8, 256, 0, stream>>>(hout, CTX16, WO16, bo, XDEC, dout, d);
  }
}
